// RecurrentTorsionGraphNetv2_74543452389444
// MI455X (gfx1250) — matrix operand order repaired and checked by reading, not compiled, not run on hardware
//
#include <hip/hip_runtime.h>
#include <hip/hip_bf16.h>

typedef __attribute__((ext_vector_type(16))) _Float16 v16h;
typedef __attribute__((ext_vector_type(8)))  _Float16 v8h;
typedef __attribute__((ext_vector_type(8)))  float    v8f;
typedef __attribute__((ext_vector_type(4)))  int      v4i;

#if defined(__has_builtin)
# if __has_builtin(__builtin_amdgcn_global_load_async_to_lds_b128)
#  define HAVE_ASYNC_LDS 1
# endif
#endif
#ifndef HAVE_ASYNC_LDS
# define HAVE_ASYNC_LDS 0
#endif

#define ASG __attribute__((address_space(1)))
#define ASL __attribute__((address_space(3)))

// Problem constants (fixed by the reference)
constexpr int NN = 16384;   // nodes
constexpr int EE = 65536;   // edges
constexpr int TT = 2048;    // torsions
constexpr int AA = 6;       // actions
constexpr int KH = 128;     // edge-MLP hidden
constexpr int KSTEPS = 258; // 256 main K-steps (K=8192) + 2 bias steps (K=64)
constexpr int SC = 3;       // k-steps per shared B chunk (divides 258)
constexpr int NCHUNK = KSTEPS / SC;  // 86

// Workspace layout (bytes, all 256-aligned)
constexpr size_t OFF_OUT   = 0;                                      // [N,64] f32 node features
constexpr size_t OFF_AGGR  = OFF_OUT  + (size_t)NN*64*4;             // [N,64] f32 scatter accumulator
constexpr size_t OFF_CNT   = OFF_AGGR + (size_t)NN*64*4;             // [N]    f32 in-degree
constexpr size_t OFF_U     = OFF_CNT  + (size_t)NN*4;                // [E,128] f16 edge-MLP layer1
constexpr size_t OFF_OUTH  = OFF_U    + (size_t)EE*KH*2;             // [N,64] f16 copy of out
constexpr size_t OFF_W2B   = OFF_OUTH + (size_t)NN*64*2;             // fragment-ready B panel
constexpr size_t OFF_E     = OFF_W2B  + (size_t)KSTEPS*4*32*16*2;    // [N] f32 softmax exp
constexpr size_t OFF_SMALL = OFF_E    + (size_t)NN*4;                // small state (floats)

// small-block float indices
constexpr int SQ_QS = 0;    // q_star [128]
constexpr int SQ_HS = 128;  // set2set h [64]
constexpr int SQ_CS = 192;  // set2set c [64]
constexpr int SQ_RA = 256;  // r accumulator [64]
constexpr int SQ_Z  = 320;  // softmax denom
constexpr int SQ_LO = 384;  // lstm_out [64]

__device__ __forceinline__ float sigm(float x) { return 1.0f / (1.0f + expf(-x)); }

__device__ __forceinline__ void wait_async0() {
#if HAVE_ASYNC_LDS
# if __has_builtin(__builtin_amdgcn_s_wait_asynccnt)
  __builtin_amdgcn_s_wait_asynccnt(0);
# else
  asm volatile("s_wait_asynccnt 0x0" ::: "memory");
# endif
#endif
}

// ---------------- utility ----------------
__global__ void k_zero(float* p, int n) {
  int i = blockIdx.x * blockDim.x + threadIdx.x;
  for (; i < n; i += gridDim.x * blockDim.x) p[i] = 0.0f;
}

// out = relu(x @ W_lin0.T + b), also f16 copy
__global__ void k_lin0(const float* __restrict__ x, const float* __restrict__ W,
                       const float* __restrict__ b, float* __restrict__ out,
                       _Float16* __restrict__ outh) {
  int idx = blockIdx.x * blockDim.x + threadIdx.x;     // N*64 threads
  int n = idx >> 6, f = idx & 63;
  float v = b[f] + x[n*3+0]*W[f*3+0] + x[n*3+1]*W[f*3+1] + x[n*3+2]*W[f*3+2];
  v = v > 0.0f ? v : 0.0f;
  out[idx] = v;
  outh[idx] = (_Float16)v;
}

// u = relu(edge_attr @ W_e1.T + b_e1) in f16
__global__ void k_edge1(const float* __restrict__ ea, const float* __restrict__ W,
                        const float* __restrict__ b, _Float16* __restrict__ u) {
  int idx = blockIdx.x * blockDim.x + threadIdx.x;     // E*128 threads
  int e = idx >> 7, j = idx & 127;
  float acc = b[j];
  #pragma unroll
  for (int c = 0; c < 7; ++c) acc += ea[e*7+c] * W[j*7+c];
  u[idx] = (_Float16)(acc > 0.0f ? acc : 0.0f);
}

// Build fragment-ready B panel from W_e2/b_e2.
//   element i of lane L, n-tile nt, k-step s:
//   K-row klocal = 16*(L>>4)+i, column f = nt*16+(L&15)
//   s <  256 : kglob = 32s+klocal ; W_e2[(d*64+f)*128 + k], k=kglob>>6, d=kglob&63
//   s >= 256 : bias rows, dglob = 32(s-256)+klocal ; b_e2[dglob*64+f]
__global__ void k_w2b(const float* __restrict__ W2, const float* __restrict__ b2,
                      _Float16* __restrict__ w2b) {
  int idx = blockIdx.x * blockDim.x + threadIdx.x;     // 258*4*32*16 threads
  int i    = idx & 15;
  int lane = (idx >> 4) & 31;
  int nt   = (idx >> 9) & 3;
  int s    = idx >> 11;
  int klocal = 16 * (lane >> 4) + i;
  int f = nt * 16 + (lane & 15);
  float v;
  if (s < 256) {
    int kglob = s * 32 + klocal;
    int k = kglob >> 6, d = kglob & 63;
    v = W2[(size_t)(d*64 + f)*128 + k];
  } else {
    int dglob = (s - 256) * 32 + klocal;
    v = b2[dglob*64 + f];
  }
  w2b[idx] = (_Float16)v;
}

__global__ void k_degree(const int* __restrict__ ei, float* __restrict__ cnt) {
  int e = blockIdx.x * blockDim.x + threadIdx.x;
  if (e < EE) atomicAdd(&cnt[ei[EE + e]], 1.0f);
}

// ---------------- WMMA message kernel ----------------
// msg[e,f] = sum_{k,d} u[e,k]*h[e,d]*W2[k,d,f] + sum_d h[e,d]*B2[d,f]
// as z[E, 8256] @ B[8256, 64] with z built on the fly.
// 8 waves/block, each wave owns a 16-edge tile; the B panel is streamed
// through LDS in double-buffered 12KB chunks shared by all 8 waves
// (8x L2-traffic reduction), filled with async-to-LDS loads when available.
__global__ void __launch_bounds__(256) k_msg(const _Float16* __restrict__ u,
                                             const _Float16* __restrict__ outh,
                                             const _Float16* __restrict__ w2b,
                                             const int* __restrict__ ei,
                                             float* __restrict__ aggr) {
  __shared__ __align__(32) _Float16 smU[8 * 2048];          // per-wave u tiles (32 KB)
  __shared__ __align__(32) _Float16 smB[2 * SC * 2048];     // B double buffer (24 KB)
  int w    = threadIdx.x >> 5;
  int lane = threadIdx.x & 31;
  int hi   = lane >> 4;
  int M    = lane & 15;
  int tile = blockIdx.x * 8 + w;                            // 4096 tiles of 16 edges
  int e0   = tile * 16;

  _Float16* up = smU + w * 2048;                            // [16][128]

  // stage u tile (contiguous 4KB per wave)
  {
    const unsigned long long* src = (const unsigned long long*)(u + (size_t)e0 * KH);
    unsigned long long* dst = (unsigned long long*)up;
    for (int q = lane; q < 512; q += 32) dst[q] = src[q];
  }
  // gather this lane's slice of the source-node h row into registers:
  // needs halves [hi*16,+16), [32+hi*16,+16)
  int srcn = ei[e0 + M];
  const v8h* hsrc = (const v8h*)(outh + (size_t)srcn * 64);
  v8h ha = hsrc[2 * hi];
  v8h hb = hsrc[2 * hi + 1];
  v8h hc = hsrc[2 * hi + 4];
  v8h hd = hsrc[2 * hi + 5];

  // prime B chunk 0
  {
    const _Float16* gsrc = w2b;
#if HAVE_ASYNC_LDS
    #pragma unroll
    for (int q = 0; q < SC; ++q) {
      size_t off = (size_t)(q * 256 + threadIdx.x) * 8;     // halves (16B granules)
      __builtin_amdgcn_global_load_async_to_lds_b128((ASG v4i*)(gsrc + off),
                                                     (ASL v4i*)(smB + off), 0, 0);
    }
    wait_async0();
#else
    const unsigned long long* s8 = (const unsigned long long*)gsrc;
    unsigned long long* d8 = (unsigned long long*)smB;
    #pragma unroll
    for (int q = 0; q < SC; ++q) {
      int t2 = (q * 256 + threadIdx.x) * 2;
      d8[t2] = s8[t2]; d8[t2 + 1] = s8[t2 + 1];
    }
#endif
  }
  __syncthreads();

  v8f acc0 = {}, acc1 = {}, acc2 = {}, acc3 = {};

  for (int c = 0; c < NCHUNK; ++c) {
    // prefetch chunk c+1 into the other buffer (its previous contents,
    // chunk c-1, were released by the barrier at the end of iteration c-1)
    if (c + 1 < NCHUNK) {
      const _Float16* gsrc = w2b + (size_t)(c + 1) * (SC * 2048);
      _Float16* ldst = smB + ((c + 1) & 1) * (SC * 2048);
#if HAVE_ASYNC_LDS
      #pragma unroll
      for (int q = 0; q < SC; ++q) {
        size_t off = (size_t)(q * 256 + threadIdx.x) * 8;
        __builtin_amdgcn_global_load_async_to_lds_b128((ASG v4i*)(gsrc + off),
                                                       (ASL v4i*)(ldst + off), 0, 0);
      }
#else
      const unsigned long long* s8 = (const unsigned long long*)gsrc;
      unsigned long long* d8 = (unsigned long long*)ldst;
      #pragma unroll
      for (int q = 0; q < SC; ++q) {
        int t2 = (q * 256 + threadIdx.x) * 2;
        d8[t2] = s8[t2]; d8[t2 + 1] = s8[t2 + 1];
      }
#endif
    }

    // process SC k-steps from the current buffer
    const v16h* bl = (const v16h*)(smB + (c & 1) * (SC * 2048));
    #pragma unroll
    for (int sl = 0; sl < SC; ++sl) {
      int s = c * SC + sl;
      _Float16 uval = (s < 256) ? up[M * 128 + (s >> 1)] : (_Float16)1.0f;
      bool odd = (s & 1) != 0;
      v8h p0 = odd ? hc : ha;
      v8h p1 = odd ? hd : hb;
      v16h a;
      #pragma unroll
      for (int i = 0; i < 8; ++i) { a[i] = uval * p0[i]; a[8 + i] = uval * p1[i]; }
      int fb = (sl * 4) * 32 + lane;
      acc0 = __builtin_amdgcn_wmma_f32_16x16x32_f16(false, a, false, bl[fb     ], (short)0, acc0, false, false);
      acc1 = __builtin_amdgcn_wmma_f32_16x16x32_f16(false, a, false, bl[fb + 32], (short)0, acc1, false, false);
      acc2 = __builtin_amdgcn_wmma_f32_16x16x32_f16(false, a, false, bl[fb + 64], (short)0, acc2, false, false);
      acc3 = __builtin_amdgcn_wmma_f32_16x16x32_f16(false, a, false, bl[fb + 96], (short)0, acc3, false, false);
    }

    wait_async0();        // own async fills of chunk c+1 are in LDS
    __syncthreads();      // all waves done reading chunk c and done filling c+1
  }

  // C layout: vgpr r -> M = r + 8*hi, lane&15 -> N. Scatter-add to aggr[tgt].
  int tg[8];
  #pragma unroll
  for (int r = 0; r < 8; ++r) tg[r] = ei[EE + e0 + r + 8*hi];
  int cc = lane & 15;
  #pragma unroll
  for (int r = 0; r < 8; ++r) {
    float* row = aggr + (size_t)tg[r] * 64;
    atomicAdd(row +      cc, acc0[r]);
    atomicAdd(row + 16 + cc, acc1[r]);
    atomicAdd(row + 32 + cc, acc2[r]);
    atomicAdd(row + 48 + cc, acc3[r]);
  }
}

// ---------------- combine + GRU (one block per node, 64 threads) ----------------
__global__ void __launch_bounds__(64) k_combine(float* __restrict__ out, _Float16* __restrict__ outh,
                        const float* __restrict__ aggr, const float* __restrict__ cnt,
                        const float* __restrict__ Wr, const float* __restrict__ br,
                        const float* __restrict__ Wih, const float* __restrict__ Whh,
                        const float* __restrict__ bih, const float* __restrict__ bhh) {
  __shared__ float row[64], mv[64];
  int n = blockIdx.x, f = threadIdx.x;
  row[f] = out[(size_t)n*64 + f];
  __syncthreads();
  float acc = br[f];
  #pragma unroll 4
  for (int d = 0; d < 64; ++d) acc += row[d] * Wr[d*64 + f];
  float cdeg = cnt[n]; cdeg = cdeg < 1.0f ? 1.0f : cdeg;
  float m = aggr[(size_t)n*64 + f] / cdeg + acc;
  m = m > 0.0f ? m : 0.0f;
  mv[f] = m;
  __syncthreads();
  float gir = bih[f], giz = bih[64+f], gin = bih[128+f];
  float ghr = bhh[f], ghz = bhh[64+f], ghn = bhh[128+f];
  #pragma unroll 4
  for (int d = 0; d < 64; ++d) {
    float md = mv[d], hd = row[d];
    gir += md * Wih[f*64       + d];
    giz += md * Wih[(64+f)*64  + d];
    gin += md * Wih[(128+f)*64 + d];
    ghr += hd * Whh[f*64       + d];
    ghz += hd * Whh[(64+f)*64  + d];
    ghn += hd * Whh[(128+f)*64 + d];
  }
  float r = sigm(gir + ghr);
  float z = sigm(giz + ghz);
  float ng = tanhf(gin + r * ghn);
  float hnew = (1.0f - z) * ng + z * row[f];
  out[(size_t)n*64 + f]  = hnew;
  outh[(size_t)n*64 + f] = (_Float16)hnew;
}

// ---------------- Set2Set ----------------
__global__ void __launch_bounds__(64) k_s2s_cell(float* __restrict__ sm,
                         const float* __restrict__ Wih, const float* __restrict__ Whh,
                         const float* __restrict__ bih, const float* __restrict__ bhh) {
  __shared__ float q[128], h[64];
  int f = threadIdx.x;
  q[f] = sm[SQ_QS + f]; q[64+f] = sm[SQ_QS + 64 + f]; h[f] = sm[SQ_HS + f];
  __syncthreads();
  float g[4];
  #pragma unroll
  for (int gg = 0; gg < 4; ++gg) {
    int j = gg*64 + f;
    float a = bih[j] + bhh[j];
    for (int k = 0; k < 128; ++k) a += q[k] * Wih[j*128 + k];
    for (int k = 0; k < 64;  ++k) a += h[k] * Whh[j*64  + k];
    g[gg] = a;
  }
  float cs = sigm(g[1]) * sm[SQ_CS + f] + sigm(g[0]) * tanhf(g[2]);
  float hs = sigm(g[3]) * tanhf(cs);
  sm[SQ_HS + f] = hs;
  sm[SQ_CS + f] = cs;
  sm[SQ_RA + f] = 0.0f;
  if (f == 0) sm[SQ_Z] = 0.0f;
}

__global__ void k_s2s_score(const float* __restrict__ out, const float* __restrict__ sm,
                            float* __restrict__ ebuf, float* __restrict__ smw) {
  __shared__ float hs[64];
  __shared__ float red[256];
  int t = threadIdx.x;
  int n = blockIdx.x * 256 + t;
  if (t < 64) hs[t] = sm[SQ_HS + t];
  __syncthreads();
  float s = 0.0f;
  #pragma unroll 4
  for (int d = 0; d < 64; ++d) s += out[(size_t)n*64 + d] * hs[d];
  float ev = expf(s);
  ebuf[n] = ev;
  red[t] = ev;
  __syncthreads();
  for (int off = 128; off > 0; off >>= 1) {
    if (t < off) red[t] += red[t + off];
    __syncthreads();
  }
  if (t == 0) atomicAdd(&smw[SQ_Z], red[0]);
}

__global__ void k_s2s_weight(const float* __restrict__ out, const float* __restrict__ ebuf,
                             float* __restrict__ sm) {
  __shared__ float pr[64];
  int t = threadIdx.x;
  int n0 = blockIdx.x * 256;
  int g = t >> 6, f = t & 63;
  if (t < 64) pr[t] = 0.0f;
  __syncthreads();
  float acc = 0.0f;
  for (int nn = g; nn < 256; nn += 4)
    acc += ebuf[n0 + nn] * out[(size_t)(n0 + nn)*64 + f];
  atomicAdd(&pr[f], acc);
  __syncthreads();
  if (t < 64) atomicAdd(&sm[SQ_RA + t], pr[t]);
}

__global__ void k_s2s_finish(float* __restrict__ sm) {
  int t = threadIdx.x;  // 128 threads
  if (t < 64) sm[SQ_QS + t] = sm[SQ_HS + t];
  else        sm[SQ_QS + t] = sm[SQ_RA + (t - 64)] / sm[SQ_Z];
}

// ---------------- memory LSTM + value head ----------------
__global__ void __launch_bounds__(64) k_mem(const float* __restrict__ hx, const float* __restrict__ cx,
                    float* __restrict__ sm,
                    const float* __restrict__ Wih, const float* __restrict__ Whh,
                    const float* __restrict__ bih, const float* __restrict__ bhh,
                    const float* __restrict__ W3, const float* __restrict__ b3,
                    float* __restrict__ dout) {
  __shared__ float pool[128], hh[64], hn[64];
  int f = threadIdx.x;
  pool[f] = sm[SQ_QS + f]; pool[64+f] = sm[SQ_QS + 64 + f]; hh[f] = hx[f];
  __syncthreads();
  float g[4];
  #pragma unroll
  for (int gg = 0; gg < 4; ++gg) {
    int j = gg*64 + f;
    float a = bih[j] + bhh[j];
    for (int k = 0; k < 128; ++k) a += pool[k] * Wih[j*128 + k];
    for (int k = 0; k < 64;  ++k) a += hh[k]  * Whh[j*64  + k];
    g[gg] = a;
  }
  float cs = sigm(g[1]) * cx[f] + sigm(g[0]) * tanhf(g[2]);
  float hs = sigm(g[3]) * tanhf(cs);
  hn[f] = hs;
  dout[TT*AA + 1 + f]      = hs;   // hx_new
  dout[TT*AA + 1 + 64 + f] = cs;   // cx_new
  sm[SQ_LO + f] = hs;
  __syncthreads();
  if (f == 0) {
    float v = b3[0];
    for (int k = 0; k < 128; ++k) v += pool[k] * W3[k];
    for (int k = 0; k < 64;  ++k) v += hn[k]   * W3[128 + k];
    dout[TT*AA] = v;
  }
}

// ---------------- torsion head (one block per torsion) ----------------
__global__ void __launch_bounds__(64) k_torsion(const float* __restrict__ out, const int* __restrict__ nonring,
                        const float* __restrict__ sm,
                        const float* __restrict__ W1, const float* __restrict__ b1,
                        const float* __restrict__ W2, const float* __restrict__ b2,
                        float* __restrict__ dout) {
  __shared__ float pf[320], hid[64];
  int t = blockIdx.x, f = threadIdx.x;
  // faithful torch view: tfeat[t,j] = out[nonring.flat[(j*T+t)>>6], (j*T+t)&63]
  for (int j = f; j < 256; j += 64) {
    int idx = j * TT + t;
    int node = nonring[idx >> 6];
    pf[j] = out[(size_t)node*64 + (idx & 63)];
  }
  for (int j = 256 + f; j < 320; j += 64) pf[j] = sm[SQ_LO + (j - 256)];
  __syncthreads();
  float a = b1[f];
  #pragma unroll 4
  for (int j = 0; j < 320; ++j) a += pf[j] * W1[f*320 + j];
  hid[f] = a > 0.0f ? a : 0.0f;
  __syncthreads();
  if (f < AA) {
    float l = b2[f];
    for (int d = 0; d < 64; ++d) l += hid[d] * W2[f*64 + d];
    dout[t*AA + f] = l;
  }
}

// ---------------- host launcher ----------------
extern "C" void kernel_launch(void* const* d_in, const int* in_sizes, int n_in,
                              void* d_out, int out_size, void* d_ws, size_t ws_size,
                              hipStream_t stream) {
  const float* x         = (const float*)d_in[0];
  const float* edge_attr = (const float*)d_in[1];
  const int*   edge_idx  = (const int*)  d_in[2];
  const int*   nonring   = (const int*)  d_in[3];
  const float* hx        = (const float*)d_in[4];
  const float* cx        = (const float*)d_in[5];
  const float* W_lin0 = (const float*)d_in[6];  const float* b_lin0 = (const float*)d_in[7];
  const float* W_e1   = (const float*)d_in[8];  const float* b_e1   = (const float*)d_in[9];
  const float* W_e2   = (const float*)d_in[10]; const float* b_e2   = (const float*)d_in[11];
  const float* conv_root = (const float*)d_in[12]; const float* conv_bias = (const float*)d_in[13];
  const float* gru_Wih = (const float*)d_in[14]; const float* gru_Whh = (const float*)d_in[15];
  const float* gru_bih = (const float*)d_in[16]; const float* gru_bhh = (const float*)d_in[17];
  const float* s2s_Wih = (const float*)d_in[18]; const float* s2s_Whh = (const float*)d_in[19];
  const float* s2s_bih = (const float*)d_in[20]; const float* s2s_bhh = (const float*)d_in[21];
  const float* mem_Wih = (const float*)d_in[22]; const float* mem_Whh = (const float*)d_in[23];
  const float* mem_bih = (const float*)d_in[24]; const float* mem_bhh = (const float*)d_in[25];
  const float* W_lin1 = (const float*)d_in[26]; const float* b_lin1 = (const float*)d_in[27];
  const float* W_lin2 = (const float*)d_in[28]; const float* b_lin2 = (const float*)d_in[29];
  const float* W_lin3 = (const float*)d_in[30]; const float* b_lin3 = (const float*)d_in[31];

  char* wsb = (char*)d_ws;
  float*    out  = (float*)   (wsb + OFF_OUT);
  float*    aggr = (float*)   (wsb + OFF_AGGR);
  float*    cnt  = (float*)   (wsb + OFF_CNT);
  _Float16* u    = (_Float16*)(wsb + OFF_U);
  _Float16* outh = (_Float16*)(wsb + OFF_OUTH);
  _Float16* w2b  = (_Float16*)(wsb + OFF_W2B);
  float*    ebuf = (float*)   (wsb + OFF_E);
  float*    smll = (float*)   (wsb + OFF_SMALL);
  float*    dout = (float*)d_out;

  // Prologue
  k_lin0 <<<NN*64/256, 256, 0, stream>>>(x, W_lin0, b_lin0, out, outh);
  k_edge1<<<EE*KH/256, 256, 0, stream>>>(edge_attr, W_e1, b_e1, u);
  k_w2b  <<<KSTEPS*4*32*16/256, 256, 0, stream>>>(W_e2, b_e2, w2b);
  k_zero <<<64, 256, 0, stream>>>(cnt, NN);
  k_zero <<<4, 256, 0, stream>>>(smll, 512);
  k_degree<<<EE/256, 256, 0, stream>>>(edge_idx, cnt);

  // 3 message-passing + GRU iterations
  for (int it = 0; it < 3; ++it) {
    k_zero<<<1024, 256, 0, stream>>>(aggr, NN*64);
    k_msg <<<512, 256, 0, stream>>>(u, outh, w2b, edge_idx, aggr);
    k_combine<<<NN, 64, 0, stream>>>(out, outh, aggr, cnt, conv_root, conv_bias,
                                     gru_Wih, gru_Whh, gru_bih, gru_bhh);
  }

  // Set2Set, 3 steps
  for (int st = 0; st < 3; ++st) {
    k_s2s_cell  <<<1, 64, 0, stream>>>(smll, s2s_Wih, s2s_Whh, s2s_bih, s2s_bhh);
    k_s2s_score <<<NN/256, 256, 0, stream>>>(out, smll, ebuf, smll);
    k_s2s_weight<<<NN/256, 256, 0, stream>>>(out, ebuf, smll);
    k_s2s_finish<<<1, 128, 0, stream>>>(smll);
  }

  // memory LSTM + value head, then torsion head
  k_mem<<<1, 64, 0, stream>>>(hx, cx, smll, mem_Wih, mem_Whh, mem_bih, mem_bhh,
                              W_lin3, b_lin3, dout);
  k_torsion<<<TT, 64, 0, stream>>>(out, nonring, smll, W_lin1, b_lin1, W_lin2, b_lin2, dout);
}
